// Attention_24318104830234
// MI455X (gfx1250) — compile-verified, matrix operand order checked by reading, not run on hardware
//
#include <hip/hip_runtime.h>

typedef __attribute__((ext_vector_type(16))) _Float16 v16h;
typedef __attribute__((ext_vector_type(8)))  _Float16 v8h;
typedef __attribute__((ext_vector_type(8)))  float    v8f;

#define WMMA(a, b, c) __builtin_amdgcn_wmma_f32_16x16x32_f16( \
    false, (a), false, (b), (short)0, (c), false, false)

// ---------------- problem constants ----------------
constexpr int B_   = 16;
constexpr int NTOK = 785;           // 1 global + 28*28 local tokens
constexpr int C_   = 768;
constexpr int H_   = 12;
constexpr int D_   = 64;
constexpr int M_   = B_ * NTOK;     // 12560 rows
constexpr int P3_  = 3 * C_;        // 2304 qkv cols
constexpr int NPAD = 800;           // padded token count (multiple of 32)
constexpr int KSTEPS = NPAD / 32;   // 25 PV K-steps
constexpr float SCALE_ = 0.125f;    // 64^-0.5

// ---------------- fragment helpers ----------------
// 16-bit operand layout (ISA 7.12.2): lane holds 16 K-values of one row/col:
//   elements 0..7  <- K = k0 + hi8 + 0..7
//   elements 8..15 <- K = k0 + 16 + hi8 + 0..7        (hi8 = 8*(lane>=16))
__device__ __forceinline__ v16h frag_zero() { v16h z = {}; return z; }

__device__ __forceinline__ v16h frag_f16(const _Float16* p, int hi8) {
    v8h c0 = *(const v8h*)(p + hi8);
    v8h c1 = *(const v8h*)(p + 16 + hi8);
    return __builtin_shufflevector(c0, c1, 0,1,2,3,4,5,6,7,8,9,10,11,12,13,14,15);
}

// fp32 source -> (hi, lo) f16 split fragments
__device__ __forceinline__ void frag_f32_split(const float* p, int hi8,
                                               v16h& fh, v16h& fl) {
#pragma unroll
    for (int i = 0; i < 8; ++i) {
        float a = p[hi8 + i];
        _Float16 h = (_Float16)a;
        fh[i] = h; fl[i] = (_Float16)(a - (float)h);
        float b = p[16 + hi8 + i];
        _Float16 h2 = (_Float16)b;
        fh[8 + i] = h2; fl[8 + i] = (_Float16)(b - (float)h2);
    }
}

__device__ __forceinline__ void split_store(float v, _Float16* dh, _Float16* dl,
                                            size_t idx) {
    _Float16 h = (_Float16)v;
    dh[idx] = h;
    dl[idx] = (_Float16)(v - (float)h);
}

// ---------------- kernel 0: fp32 -> split f16 ----------------
__global__ __launch_bounds__(256) void k_convert(const float* __restrict__ src,
                                                 _Float16* __restrict__ dh,
                                                 _Float16* __restrict__ dl, int n) {
    int stride = gridDim.x * blockDim.x;
    for (int i = blockIdx.x * blockDim.x + threadIdx.x; i < n; i += stride) {
        float a = src[i];
        _Float16 h = (_Float16)a;
        dh[i] = h;
        dl[i] = (_Float16)(a - (float)h);
    }
}

__global__ __launch_bounds__(256) void k_zero16(_Float16* __restrict__ p, int n) {
    int stride = gridDim.x * blockDim.x;
    for (int i = blockIdx.x * blockDim.x + threadIdx.x; i < n; i += stride)
        p[i] = (_Float16)0.0f;
}

// ---------------- kernel 1: QKV GEMM ----------------
__global__ __launch_bounds__(32) void k_qkv(
    const float* __restrict__ x,
    const _Float16* __restrict__ wh, const _Float16* __restrict__ wl,
    _Float16* __restrict__ qh, _Float16* __restrict__ ql,
    _Float16* __restrict__ kh, _Float16* __restrict__ kl,
    _Float16* __restrict__ vth, _Float16* __restrict__ vtl) {
    const int lane = threadIdx.x;
    const int lh   = lane & 15;
    const int hv   = lane >> 4;
    const int hi8  = hv << 3;
    const int pt0  = blockIdx.x * 32;   // col tile base (always in range)
    const int mt0  = blockIdx.y * 32;   // row tile base

    const int rowA0 = mt0 + lh, rowA1 = mt0 + 16 + lh;
    const bool va0 = rowA0 < M_, va1 = rowA1 < M_;
    const float* a0p = x + (size_t)rowA0 * C_;
    const float* a1p = x + (size_t)rowA1 * C_;
    const _Float16* b0h = wh + (size_t)(pt0 + lh) * C_;
    const _Float16* b0l = wl + (size_t)(pt0 + lh) * C_;
    const _Float16* b1h = wh + (size_t)(pt0 + 16 + lh) * C_;
    const _Float16* b1l = wl + (size_t)(pt0 + 16 + lh) * C_;

    v8f acc00 = {}, acc01 = {}, acc10 = {}, acc11 = {};
    for (int k0 = 0; k0 < C_; k0 += 32) {
        // unconditional speculative prefetch of the next K-step (silently
        // dropped on translation failure -> safe past the buffer end; no
        // branches, no EXEC juggling in the inner loop)
        __builtin_prefetch(a0p + k0 + 32, 0, 1);
        __builtin_prefetch(a1p + k0 + 32, 0, 1);
        __builtin_prefetch(b0h + k0 + 32, 0, 1);
        __builtin_prefetch(b1h + k0 + 32, 0, 1);

        v16h a0H, a0L, a1H, a1L;
        if (va0) frag_f32_split(a0p + k0, hi8, a0H, a0L);
        else { a0H = frag_zero(); a0L = frag_zero(); }
        if (va1) frag_f32_split(a1p + k0, hi8, a1H, a1L);
        else { a1H = frag_zero(); a1L = frag_zero(); }
        v16h bH0 = frag_f16(b0h + k0, hi8), bL0 = frag_f16(b0l + k0, hi8);
        v16h bH1 = frag_f16(b1h + k0, hi8), bL1 = frag_f16(b1l + k0, hi8);

        acc00 = WMMA(a0H, bH0, acc00); acc00 = WMMA(a0H, bL0, acc00); acc00 = WMMA(a0L, bH0, acc00);
        acc01 = WMMA(a0H, bH1, acc01); acc01 = WMMA(a0H, bL1, acc01); acc01 = WMMA(a0L, bH1, acc01);
        acc10 = WMMA(a1H, bH0, acc10); acc10 = WMMA(a1H, bL0, acc10); acc10 = WMMA(a1L, bH0, acc10);
        acc11 = WMMA(a1H, bH1, acc11); acc11 = WMMA(a1H, bL1, acc11); acc11 = WMMA(a1L, bH1, acc11);
    }

    // scatter stores: col -> (o, head, t)
#pragma unroll
    for (int sub = 0; sub < 2; ++sub) {
#pragma unroll
        for (int ct = 0; ct < 2; ++ct) {
            const v8f& acc = sub == 0 ? (ct == 0 ? acc00 : acc01)
                                      : (ct == 0 ? acc10 : acc11);
            int col = pt0 + ct * 16 + lh;
            int o   = col / C_;
            int rem = col - o * C_;
            int hd  = rem >> 6;
            int t   = rem & 63;
#pragma unroll
            for (int r = 0; r < 8; ++r) {
                int row = mt0 + sub * 16 + r + 8 * hv;
                if (row >= M_) continue;
                int b = row / NTOK;
                int n = row - b * NTOK;
                size_t bh = (size_t)(b * H_ + hd);
                float v = acc[r];
                if (o == 0)      split_store(v, qh, ql, (bh * NTOK + n) * D_ + t);
                else if (o == 1) split_store(v, kh, kl, (bh * NTOK + n) * D_ + t);
                else             split_store(v, vth, vtl, (bh * D_ + t) * NPAD + n);
            }
        }
    }
}

// ---------------- kernel 2: attention ----------------
__global__ __launch_bounds__(32) void k_attn(
    const _Float16* __restrict__ qh, const _Float16* __restrict__ ql,
    const _Float16* __restrict__ kh, const _Float16* __restrict__ kl,
    const _Float16* __restrict__ vth, const _Float16* __restrict__ vtl,
    const float* __restrict__ rel_table, const float* __restrict__ g2l,
    const float* __restrict__ g2g,
    _Float16* __restrict__ aoh, _Float16* __restrict__ aol) {
    __shared__ float sc[16][NPAD];           // 51.2 KB scores
    __shared__ _Float16 sv[2][2][64][32];    // 16 KB double-buffered Vt tile

    const int lane = threadIdx.x;
    const int lh   = lane & 15;
    const int hv   = lane >> 4;
    const int hi8  = hv << 3;
    const int n0   = blockIdx.x * 16;
    const int hd   = blockIdx.y;
    const int b    = blockIdx.z;
    const size_t bh = (size_t)(b * H_ + hd);

    // Q fragments (A side), loaded once
    const int qrow = n0 + lh;
    const bool qv = qrow < NTOK;
    const _Float16* qph = qh + (bh * NTOK + qrow) * D_;
    const _Float16* qpl = ql + (bh * NTOK + qrow) * D_;
    v16h qfh[2], qfl[2];
#pragma unroll
    for (int ks = 0; ks < 2; ++ks) {
        qfh[ks] = qv ? frag_f16(qph + ks * 32, hi8) : frag_zero();
        qfl[ks] = qv ? frag_f16(qpl + ks * 32, hi8) : frag_zero();
    }

    // ---- scores: S = (Q Kt) * scale + bias ----
    for (int mt = 0; mt < NPAD / 16; ++mt) {
        int m0 = mt * 16;
        int kcol = m0 + lh;
        bool kv = kcol < NTOK;
        const _Float16* kph = kh + (bh * NTOK + (kv ? kcol : 0)) * D_;
        const _Float16* kpl = kl + (bh * NTOK + (kv ? kcol : 0)) * D_;

        v8f acc = {};
#pragma unroll
        for (int ks = 0; ks < 2; ++ks) {
            v16h kH = kv ? frag_f16(kph + ks * 32, hi8) : frag_zero();
            v16h kL = kv ? frag_f16(kpl + ks * 32, hi8) : frag_zero();
            acc = WMMA(qfh[ks], kH, acc);
            acc = WMMA(qfh[ks], kL, acc);
            acc = WMMA(qfl[ks], kH, acc);
        }

        int m = m0 + lh;           // key-token index for this lane (C/D col)
#pragma unroll
        for (int r = 0; r < 8; ++r) {
            int qr = n0 + r + 8 * hv;   // query-token index (C/D row)
            if (m < NTOK) {
                float bias;
                if (qr == 0) {
                    bias = (m == 0) ? g2g[hd] : g2l[hd];          // g2g[h,0,0] / g2l[0,h,0]
                } else if (m == 0) {
                    bias = g2l[H_ + hd];                          // g2l[1,h,0]
                } else if (qr < NTOK) {
                    int i = qr - 1, j = m - 1;
                    int dx = (i / 28) - (j / 28) + 27;
                    int dy = (i % 28) - (j % 28) + 27;
                    bias = rel_table[(dx * 55 + dy) * H_ + hd];
                } else {
                    bias = 0.0f;
                }
                sc[r + 8 * hv][m] = acc[r] * SCALE_ + bias;
            }
        }
    }
    // pad columns to -inf
    for (int c = NTOK + lane; c < NPAD; c += 32)
#pragma unroll
        for (int r = 0; r < 16; ++r) sc[r][c] = -1e30f;
    __syncthreads();

    // ---- exact f32 softmax per row (wave32 shuffle reductions) ----
    for (int r = 0; r < 16; ++r) {
        float mx = -1e30f;
        for (int c = lane; c < NPAD; c += 32) mx = fmaxf(mx, sc[r][c]);
#pragma unroll
        for (int off = 16; off; off >>= 1) mx = fmaxf(mx, __shfl_xor(mx, off, 32));
        float sum = 0.0f;
        for (int c = lane; c < NPAD; c += 32) {
            float e = __expf(sc[r][c] - mx);
            sc[r][c] = e;
            sum += e;
        }
#pragma unroll
        for (int off = 16; off; off >>= 1) sum += __shfl_xor(sum, off, 32);
        float rs = 1.0f / sum;
        for (int c = lane; c < NPAD; c += 32) sc[r][c] *= rs;
    }
    __syncthreads();

    // ---- out = P @ V ----
    // Vt tiles (8 KB per K-step, fully padded => no predication) are DMA'd
    // into LDS with GLOBAL_LOAD_ASYNC_TO_LDS_B128 (ASYNCcnt) and double
    // buffered: issue fill(ks+1), wait asynccnt<=16 so fill(ks) has landed.
    const unsigned svoff = (unsigned)(uintptr_t)&sv[0][0][0][0];
    const _Float16* vbh = vth + bh * D_ * NPAD;
    const _Float16* vbl = vtl + bh * D_ * NPAD;

    auto fill_v = [&](int buf, int ks) {
        unsigned dstbase = svoff + (unsigned)buf * (2u * 64u * 32u * 2u);
#pragma unroll
        for (int i = 0; i < 16; ++i) {
            int c    = lane + 32 * i;      // 0..511 : 16B chunks of the tile
            int half = c >> 8;             // 0 = hi array, 1 = lo array
            int cc   = c & 255;
            int d    = cc >> 2;            // Vt row (d index)
            int col  = (cc & 3) * 8;       // half-offset within 32-wide chunk
            const _Float16* src = (half ? vbl : vbh) + (size_t)d * NPAD + ks * 32 + col;
            unsigned dst = dstbase + (unsigned)(((half * 64 + d) * 32 + col) * 2);
            asm volatile("global_load_async_to_lds_b128 %0, %1, off"
                         :: "v"(dst), "v"(src) : "memory");
        }
    };

    v8f oacc[4] = {{}, {}, {}, {}};
    fill_v(0, 0);
    int buf = 0;
    for (int ks = 0; ks < KSTEPS; ++ks) {
        if (ks + 1 < KSTEPS) {
            fill_v(buf ^ 1, ks + 1);
            asm volatile("s_wait_asynccnt 0x10" ::: "memory");  // fill(ks) done
        } else {
            asm volatile("s_wait_asynccnt 0x0" ::: "memory");
        }

        v16h pfh, pfl;
        frag_f32_split(&sc[lh][ks * 32], hi8, pfh, pfl);
#pragma unroll
        for (int dt = 0; dt < 4; ++dt) {
            int d = dt * 16 + lh;
            v16h vH = frag_f16(&sv[buf][0][d][0], hi8);
            v16h vL = frag_f16(&sv[buf][1][d][0], hi8);
            oacc[dt] = WMMA(pfh, vH, oacc[dt]);
            oacc[dt] = WMMA(pfh, vL, oacc[dt]);
            oacc[dt] = WMMA(pfl, vH, oacc[dt]);
        }
        buf ^= 1;
    }
#pragma unroll
    for (int dt = 0; dt < 4; ++dt) {
#pragma unroll
        for (int r = 0; r < 8; ++r) {
            int row = n0 + r + 8 * hv;
            if (row < NTOK) {
                size_t idx = ((size_t)(b * NTOK + row)) * C_ + hd * 64 + dt * 16 + lh;
                split_store(oacc[dt][r], aoh, aol, idx);
            }
        }
    }
}

// ---------------- kernel 3: output projection ----------------
__global__ __launch_bounds__(32) void k_proj(
    const _Float16* __restrict__ ah, const _Float16* __restrict__ al,
    const _Float16* __restrict__ wh, const _Float16* __restrict__ wl,
    const float* __restrict__ bias, float* __restrict__ out) {
    const int lane = threadIdx.x;
    const int lh   = lane & 15;
    const int hv   = lane >> 4;
    const int hi8  = hv << 3;
    const int pt0  = blockIdx.x * 32;
    const int mt0  = blockIdx.y * 32;

    const int rowA0 = mt0 + lh, rowA1 = mt0 + 16 + lh;
    const bool va0 = rowA0 < M_, va1 = rowA1 < M_;
    const _Float16* a0h = ah + (size_t)(va0 ? rowA0 : 0) * C_;
    const _Float16* a0l = al + (size_t)(va0 ? rowA0 : 0) * C_;
    const _Float16* a1h = ah + (size_t)(va1 ? rowA1 : 0) * C_;
    const _Float16* a1l = al + (size_t)(va1 ? rowA1 : 0) * C_;
    const _Float16* b0h = wh + (size_t)(pt0 + lh) * C_;
    const _Float16* b0l = wl + (size_t)(pt0 + lh) * C_;
    const _Float16* b1h = wh + (size_t)(pt0 + 16 + lh) * C_;
    const _Float16* b1l = wl + (size_t)(pt0 + 16 + lh) * C_;

    v8f acc00 = {}, acc01 = {}, acc10 = {}, acc11 = {};
    for (int k0 = 0; k0 < C_; k0 += 32) {
        __builtin_prefetch(a0h + k0 + 32, 0, 1);
        __builtin_prefetch(a1h + k0 + 32, 0, 1);
        __builtin_prefetch(b0h + k0 + 32, 0, 1);
        __builtin_prefetch(b1h + k0 + 32, 0, 1);

        v16h a0H = va0 ? frag_f16(a0h + k0, hi8) : frag_zero();
        v16h a0L = va0 ? frag_f16(a0l + k0, hi8) : frag_zero();
        v16h a1H = va1 ? frag_f16(a1h + k0, hi8) : frag_zero();
        v16h a1L = va1 ? frag_f16(a1l + k0, hi8) : frag_zero();
        v16h bH0 = frag_f16(b0h + k0, hi8), bL0 = frag_f16(b0l + k0, hi8);
        v16h bH1 = frag_f16(b1h + k0, hi8), bL1 = frag_f16(b1l + k0, hi8);

        acc00 = WMMA(a0H, bH0, acc00); acc00 = WMMA(a0H, bL0, acc00); acc00 = WMMA(a0L, bH0, acc00);
        acc01 = WMMA(a0H, bH1, acc01); acc01 = WMMA(a0H, bL1, acc01); acc01 = WMMA(a0L, bH1, acc01);
        acc10 = WMMA(a1H, bH0, acc10); acc10 = WMMA(a1H, bL0, acc10); acc10 = WMMA(a1L, bH0, acc10);
        acc11 = WMMA(a1H, bH1, acc11); acc11 = WMMA(a1H, bL1, acc11); acc11 = WMMA(a1L, bH1, acc11);
    }

#pragma unroll
    for (int sub = 0; sub < 2; ++sub) {
#pragma unroll
        for (int ct = 0; ct < 2; ++ct) {
            const v8f& acc = sub == 0 ? (ct == 0 ? acc00 : acc01)
                                      : (ct == 0 ? acc10 : acc11);
            int col = pt0 + ct * 16 + lh;
            float bb = bias[col];
#pragma unroll
            for (int r = 0; r < 8; ++r) {
                int row = mt0 + sub * 16 + r + 8 * hv;
                if (row < M_) out[(size_t)row * C_ + col] = acc[r] + bb;
            }
        }
    }
}

// ---------------- host launcher ----------------
extern "C" void kernel_launch(void* const* d_in, const int* in_sizes, int n_in,
                              void* d_out, int out_size, void* d_ws, size_t ws_size,
                              hipStream_t stream) {
    const float* x         = (const float*)d_in[0];
    const float* qkv_w     = (const float*)d_in[1];
    const float* proj_w    = (const float*)d_in[2];
    const float* proj_b    = (const float*)d_in[3];
    const float* rel_table = (const float*)d_in[4];
    const float* g2l       = (const float*)d_in[5];
    const float* g2g       = (const float*)d_in[6];
    float* out = (float*)d_out;

    constexpr size_t WQ = (size_t)P3_ * C_;              // 1,769,472
    constexpr size_t PW = (size_t)C_ * C_;               //   589,824
    constexpr size_t QK = (size_t)B_ * H_ * NTOK * D_;   // 9,646,080
    constexpr size_t VT = (size_t)B_ * H_ * D_ * NPAD;   // 9,830,400
    constexpr size_t AO = (size_t)M_ * C_;               // 9,646,080

    char* p = (char*)d_ws;
    auto take = [&](size_t elems) {
        _Float16* r = (_Float16*)p;
        p += (elems * sizeof(_Float16) + 255) & ~(size_t)255;
        return r;
    };
    _Float16* wqh = take(WQ); _Float16* wql = take(WQ);
    _Float16* pwh = take(PW); _Float16* pwl = take(PW);
    _Float16* qh  = take(QK); _Float16* ql  = take(QK);
    _Float16* kh  = take(QK); _Float16* kl  = take(QK);
    _Float16* vth = take(VT); _Float16* vtl = take(VT);
    _Float16* aoh = take(AO); _Float16* aol = take(AO);

    k_convert<<<dim3(2048), dim3(256), 0, stream>>>(qkv_w, wqh, wql, (int)WQ);
    k_convert<<<dim3(1024), dim3(256), 0, stream>>>(proj_w, pwh, pwl, (int)PW);
    k_zero16 <<<dim3(2048), dim3(256), 0, stream>>>(vth, (int)VT);
    k_zero16 <<<dim3(2048), dim3(256), 0, stream>>>(vtl, (int)VT);

    k_qkv<<<dim3(P3_ / 32, (M_ + 31) / 32), dim3(32), 0, stream>>>(
        x, wqh, wql, qh, ql, kh, kl, vth, vtl);

    k_attn<<<dim3(NPAD / 16, H_, B_), dim3(32), 0, stream>>>(
        qh, ql, kh, kl, vth, vtl, rel_table, g2l, g2g, aoh, aol);

    k_proj<<<dim3(C_ / 32, (M_ + 31) / 32), dim3(32), 0, stream>>>(
        aoh, aol, pwh, pwl, proj_b, out);
}
